// MyAttentionCellWrapper_43258910605611
// MI455X (gfx1250) — compile-verified, matrix operand order checked by reading, not run on hardware
//
#include <hip/hip_runtime.h>
#include <hip/hip_bf16.h>

typedef __bf16 v16bf __attribute__((ext_vector_type(16)));
typedef float v8f  __attribute__((ext_vector_type(8)));
typedef int  v4i  __attribute__((ext_vector_type(4)));

#define BM 128
#define BN 128
#define BKK 32
#define LDK 40   // 32 + 8 pad; 80B row stride keeps 16B alignment

#if defined(__gfx1250__) && __has_builtin(__builtin_amdgcn_global_load_async_to_lds_b128)
#define USE_ASYNC_LDS 1
#else
#define USE_ASYNC_LDS 0
#endif

union Frag16 { uint4 q[2]; v16bf v; };
union Pack8  { __bf16 h[8]; uint4 q; };

__device__ __forceinline__ uint4 pack8_bf16(float4 a, float4 b) {
    Pack8 p;
    p.h[0]=(__bf16)a.x; p.h[1]=(__bf16)a.y; p.h[2]=(__bf16)a.z; p.h[3]=(__bf16)a.w;
    p.h[4]=(__bf16)b.x; p.h[5]=(__bf16)b.y; p.h[6]=(__bf16)b.z; p.h[7]=(__bf16)b.w;
    return p.q;
}

__device__ __forceinline__ void async_copy_b128(const void* gsrc, void* lds_dst) {
#if USE_ASYNC_LDS
    __builtin_amdgcn_global_load_async_to_lds_b128(
        (__attribute__((address_space(1))) v4i*)(uintptr_t)gsrc,
        (__attribute__((address_space(3))) v4i*)(uint32_t)(uintptr_t)lds_dst,
        0, 0);
#else
    *(uint4*)lds_dst = *(const uint4*)gsrc;
#endif
}

__device__ __forceinline__ void wait_async_lds() {
#if USE_ASYNC_LDS
#if __has_builtin(__builtin_amdgcn_s_wait_asynccnt)
    __builtin_amdgcn_s_wait_asynccnt(0);
#else
    asm volatile("s_wait_asynccnt 0x0" ::: "memory");
#endif
#endif
}

// ---------------------------------------------------------------------------
// Tiled bf16 WMMA GEMM: C[M,N] = epi( [A1|A2] @ [B1;B2] + bias )
// A sources are fp32, row-contiguous with row length == their K (converted to
// bf16 in-register while staging to LDS). B sources are pre-converted bf16,
// TRANSPOSED layout [N][K_i], staged with async global->LDS (ASYNCcnt path).
// EPI 0: store fp32 C + bias
// EPI 1: store fp32 C + bias to out and out2 (second stride/offset)
// EPI 2: fused score: atomicAdd(s[row], sum_col tanh(C+y[row>>4,col])*vsum[col])
// ---------------------------------------------------------------------------
template<int EPI>
__global__ __launch_bounds__(256)
void gemm_bf16(const float* __restrict__ A1, const float* __restrict__ A2,
               int K1, int Ktot,
               const __bf16* __restrict__ B1t, const __bf16* __restrict__ B2t,
               const float* __restrict__ bias,
               float* __restrict__ out, int ldc,
               float* __restrict__ out2, int ld2, int off2,
               const float* __restrict__ yv, const float* __restrict__ vsum,
               float* __restrict__ s_out)
{
    __shared__ __bf16 Asub[BM][LDK];
    __shared__ __bf16 Bsub[BN][LDK];

    const int tid  = threadIdx.x;
    const int lane = tid & 31;
    const int wave = tid >> 5;
    const int wm   = wave & 1;    // 2 waves along M (64 rows each)
    const int wn   = wave >> 1;   // 4 waves along N (32 cols each)
    const int rowBase = blockIdx.y * BM + wm * 64;
    const int colBase = blockIdx.x * BN + wn * 32;
    const int halfsel = lane >> 4;   // K half selector per WMMA layout
    const int lidx    = lane & 15;

    v8f acc[4][2] = {};

    const int srow = tid >> 1;        // staging: 2 threads per 128 rows
    const int scol = (tid & 1) * 16;  // 16 elements each

    for (int k0 = 0; k0 < Ktot; k0 += BKK) {
        const float*  Ap; const __bf16* Bp; int ks, kstride;
        if (k0 < K1) { Ap = A1; Bp = B1t; ks = k0;      kstride = K1;        }
        else         { Ap = A2; Bp = B2t; ks = k0 - K1; kstride = Ktot - K1; }

        // stage B tile: async global->LDS, no VGPR round trip (already bf16)
        {
            const __bf16* srcb = Bp + (size_t)(blockIdx.x * BN + srow) * kstride + ks + scol;
            async_copy_b128(srcb,     &Bsub[srow][scol]);
            async_copy_b128(srcb + 8, &Bsub[srow][scol + 8]);
        }
        // stage A tile (fp32 -> bf16 in registers), 128x32
        {
            const float* src = Ap + (size_t)(blockIdx.y * BM + srow) * kstride + ks + scol;
            __builtin_prefetch(src + BKK, 0, 1);   // speculative next-tile prefetch
            float4 f0 = *(const float4*)(src + 0);
            float4 f1 = *(const float4*)(src + 4);
            float4 f2 = *(const float4*)(src + 8);
            float4 f3 = *(const float4*)(src + 12);
            *(uint4*)&Asub[srow][scol]     = pack8_bf16(f0, f1);
            *(uint4*)&Asub[srow][scol + 8] = pack8_bf16(f2, f3);
        }
        wait_async_lds();
        __syncthreads();

        Frag16 afrag[4], bfrag[2];
        #pragma unroll
        for (int mt = 0; mt < 4; ++mt) {
            const __bf16* ap = &Asub[wm * 64 + mt * 16 + lidx][halfsel * 16];
            afrag[mt].q[0] = *(const uint4*)ap;
            afrag[mt].q[1] = *(const uint4*)(ap + 8);
        }
        #pragma unroll
        for (int nt = 0; nt < 2; ++nt) {
            const __bf16* bp = &Bsub[wn * 32 + nt * 16 + lidx][halfsel * 16];
            bfrag[nt].q[0] = *(const uint4*)bp;
            bfrag[nt].q[1] = *(const uint4*)(bp + 8);
        }
        #pragma unroll
        for (int mt = 0; mt < 4; ++mt)
            #pragma unroll
            for (int nt = 0; nt < 2; ++nt)
                acc[mt][nt] = __builtin_amdgcn_wmma_f32_16x16x32_bf16(
                    false, afrag[mt].v, false, bfrag[nt].v,
                    (short)0, acc[mt][nt], false, false);

        __syncthreads();
    }

    if (EPI == 0 || EPI == 1) {
        #pragma unroll
        for (int mt = 0; mt < 4; ++mt) {
            int row0 = rowBase + mt * 16 + halfsel * 8;
            #pragma unroll
            for (int nt = 0; nt < 2; ++nt) {
                int col = colBase + nt * 16 + lidx;
                float bv = bias ? bias[col] : 0.0f;
                #pragma unroll
                for (int r = 0; r < 8; ++r) {
                    float val = acc[mt][nt][r] + bv;
                    out[(size_t)(row0 + r) * ldc + col] = val;
                    if (EPI == 1)
                        out2[(size_t)(row0 + r) * ld2 + off2 + col] = val;
                }
            }
        }
    } else { // EPI == 2: fused tanh * vsum row-reduction into scores
        #pragma unroll
        for (int mt = 0; mt < 4; ++mt) {
            int row0 = rowBase + mt * 16 + halfsel * 8;
            #pragma unroll
            for (int r = 0; r < 8; ++r) {
                int row = row0 + r;
                const float* yrow = yv + (size_t)(row >> 4) * 512;
                float part = 0.0f;
                #pragma unroll
                for (int nt = 0; nt < 2; ++nt) {
                    int col = colBase + nt * 16 + lidx;
                    float t = tanhf(acc[mt][nt][r] + yrow[col]);
                    part += t * vsum[col];
                }
                #pragma unroll
                for (int m = 8; m >= 1; m >>= 1)
                    part += __shfl_xor(part, m, 32);
                if (lidx == 0)
                    atomicAdd(&s_out[row], part);
            }
        }
    }
}

// fp32 [Koff+k][N] -> bf16 transposed [n*Ksub + k]
__global__ void conv_transpose_bf16(const float* __restrict__ src, __bf16* __restrict__ dst,
                                    int Koff, int Ksub, int N) {
    int idx = blockIdx.x * blockDim.x + threadIdx.x;
    if (idx >= Ksub * N) return;
    int n = idx / Ksub;
    int k = idx - n * Ksub;
    dst[idx] = (__bf16)src[(size_t)(Koff + k) * N + n];
}

__global__ void vsum_kernel(const float* __restrict__ v, float* __restrict__ vsum) {
    int n = blockIdx.x * blockDim.x + threadIdx.x;
    if (n >= 512) return;
    float acc = 0.0f;
    for (int s = 0; s < 512; ++s) acc += v[(size_t)s * 512 + n];
    vsum[n] = acc;
}

__global__ void lstm_elem(const float* __restrict__ gates, const float* __restrict__ cell_c,
                          float* __restrict__ c_out, float* __restrict__ h_out, int BH) {
    int i = blockIdx.x * blockDim.x + threadIdx.x;
    if (i >= BH) return;
    int b = i >> 9;
    int h = i & 511;
    const float* g = gates + (size_t)b * 2048;
    float gi = g[h], gf = g[512 + h], gg = g[1024 + h], go = g[1536 + h];
    float si = 1.0f / (1.0f + __expf(-gi));
    float sf = 1.0f / (1.0f + __expf(-gf));
    float so = 1.0f / (1.0f + __expf(-go));
    float c  = sf * cell_c[i] + si * tanhf(gg);
    c_out[i] = c;
    h_out[i] = so * tanhf(c);
}

__global__ __launch_bounds__(256)
void softmax_attend(const float* __restrict__ s, const float* __restrict__ attn_states,
                    float* __restrict__ d_attn) {
    int b = blockIdx.x;
    __shared__ float sv[16];
    if (threadIdx.x < 16) sv[threadIdx.x] = s[(size_t)b * 16 + threadIdx.x];
    __syncthreads();
    float mx = sv[0];
    #pragma unroll
    for (int l = 1; l < 16; ++l) mx = fmaxf(mx, sv[l]);
    float e[16];
    float den = 0.0f;
    #pragma unroll
    for (int l = 0; l < 16; ++l) { e[l] = __expf(sv[l] - mx); den += e[l]; }
    float inv = 1.0f / den;
    for (int c = threadIdx.x; c < 512; c += 256) {
        const float* as = attn_states + (size_t)b * 8192 + c;
        float acc = 0.0f;
        #pragma unroll
        for (int l = 0; l < 16; ++l) acc += e[l] * as[(size_t)l * 512];
        d_attn[(size_t)b * 512 + c] = acc * inv;
    }
}

__global__ void shift_states(const float* __restrict__ attn_states, float* __restrict__ nas) {
    size_t idx = (size_t)blockIdx.x * blockDim.x + threadIdx.x; // float4 units
    const size_t total = (size_t)4096 * 1920;                   // 15*512/4 per batch
    if (idx >= total) return;
    size_t b = idx / 1920;
    size_t j = idx - b * 1920;
    const float4* src = (const float4*)(attn_states + b * 8192 + 512) + j;
    float4*       dst = (float4*)(nas + b * 8192) + j;
    *dst = *src;
}

extern "C" void kernel_launch(void* const* d_in, const int* in_sizes, int n_in,
                              void* d_out, int out_size, void* d_ws, size_t ws_size,
                              hipStream_t stream) {
    (void)in_sizes; (void)n_in; (void)out_size; (void)ws_size;
    const float* inputs      = (const float*)d_in[0];
    const float* cell_c      = (const float*)d_in[1];
    const float* cell_h      = (const float*)d_in[2];
    const float* attns       = (const float*)d_in[3];
    const float* attn_states = (const float*)d_in[4];
    const float* W1          = (const float*)d_in[5];
    const float* b1          = (const float*)d_in[6];
    const float* Wx          = (const float*)d_in[7];
    const float* Wh          = (const float*)d_in[8];
    const float* b_lstm      = (const float*)d_in[9];
    const float* W3          = (const float*)d_in[10];
    const float* b3          = (const float*)d_in[11];
    const float* kmat        = (const float*)d_in[12];
    const float* vmat        = (const float*)d_in[13];
    const float* W2          = (const float*)d_in[14];
    const float* b2          = (const float*)d_in[15];

    float* dout = (float*)d_out;
    const size_t off_out = 0;
    const size_t off_c   = (size_t)4096 * 512;
    const size_t off_h   = off_c + (size_t)4096 * 512;
    const size_t off_d   = off_h + (size_t)4096 * 512;
    const size_t off_nas = off_d + (size_t)4096 * 512;

    // workspace carve-up
    char* ws = (char*)d_ws;
    size_t o = 0;
    auto alloc = [&](size_t bytes) { size_t r = o; o = (o + bytes + 255) & ~(size_t)255; return r; };
    __bf16* W1at = (__bf16*)(ws + alloc(512 * 512 * 2));
    __bf16* W1bt = (__bf16*)(ws + alloc(512 * 512 * 2));
    __bf16* Wxt  = (__bf16*)(ws + alloc((size_t)2048 * 512 * 2));
    __bf16* Wht  = (__bf16*)(ws + alloc((size_t)2048 * 512 * 2));
    __bf16* W3at = (__bf16*)(ws + alloc(512 * 512 * 2));
    __bf16* W3bt = (__bf16*)(ws + alloc(512 * 512 * 2));
    __bf16* W2at = (__bf16*)(ws + alloc(512 * 512 * 2));
    __bf16* W2bt = (__bf16*)(ws + alloc(512 * 512 * 2));
    __bf16* kt   = (__bf16*)(ws + alloc(512 * 512 * 2));
    float*  vsum = (float*)(ws + alloc(512 * 4));
    float*  x    = (float*)(ws + alloc((size_t)4096 * 512 * 4));
    float*  gts  = (float*)(ws + alloc((size_t)4096 * 2048 * 4));
    float*  ybuf = (float*)(ws + alloc((size_t)4096 * 512 * 4));
    float*  sbuf = (float*)(ws + alloc((size_t)65536 * 4));

    auto g1 = [](size_t n) { return dim3((unsigned)((n + 255) / 256)); };

    // weight convert + transpose (bf16 [N][K] panels)
    conv_transpose_bf16<<<g1(512 * 512), 256, 0, stream>>>(W1, W1at, 0, 512, 512);
    conv_transpose_bf16<<<g1(512 * 512), 256, 0, stream>>>(W1, W1bt, 512, 512, 512);
    conv_transpose_bf16<<<g1((size_t)2048 * 512), 256, 0, stream>>>(Wx, Wxt, 0, 512, 2048);
    conv_transpose_bf16<<<g1((size_t)2048 * 512), 256, 0, stream>>>(Wh, Wht, 0, 512, 2048);
    conv_transpose_bf16<<<g1(512 * 512), 256, 0, stream>>>(W3, W3at, 0, 512, 512);
    conv_transpose_bf16<<<g1(512 * 512), 256, 0, stream>>>(W3, W3bt, 512, 512, 512);
    conv_transpose_bf16<<<g1(512 * 512), 256, 0, stream>>>(W2, W2at, 0, 512, 512);
    conv_transpose_bf16<<<g1(512 * 512), 256, 0, stream>>>(W2, W2bt, 512, 512, 512);
    conv_transpose_bf16<<<g1(512 * 512), 256, 0, stream>>>(kmat, kt, 0, 512, 512);
    vsum_kernel<<<2, 256, 0, stream>>>(vmat, vsum);
    (void)hipMemsetAsync(sbuf, 0, (size_t)65536 * 4, stream);

    // x = [inputs|attns] @ W1 + b1
    gemm_bf16<0><<<dim3(4, 32), 256, 0, stream>>>(inputs, attns, 512, 1024, W1at, W1bt,
        b1, x, 512, nullptr, 0, 0, nullptr, nullptr, nullptr);
    // gates = x @ Wx + h @ Wh + b_lstm
    gemm_bf16<0><<<dim3(16, 32), 256, 0, stream>>>(x, cell_h, 512, 1024, Wxt, Wht,
        b_lstm, gts, 2048, nullptr, 0, 0, nullptr, nullptr, nullptr);
    // LSTM pointwise -> c_new, h_new (written straight into d_out)
    lstm_elem<<<g1((size_t)4096 * 512), 256, 0, stream>>>(gts, cell_c,
        dout + off_c, dout + off_h, 4096 * 512);
    // y = [c_new|h_new] @ W3 + b3
    gemm_bf16<0><<<dim3(4, 32), 256, 0, stream>>>(dout + off_c, dout + off_h, 512, 1024,
        W3at, W3bt, b3, ybuf, 512, nullptr, 0, 0, nullptr, nullptr, nullptr);
    // s[b,l] = sum_v tanh((attn_states@k)[b,l,v] + y[b,v]) * vsum[v]  (fused epilogue)
    gemm_bf16<2><<<dim3(4, 512), 256, 0, stream>>>(attn_states, nullptr, 512, 512,
        kt, nullptr, nullptr, nullptr, 0, nullptr, 0, 0, ybuf, vsum, sbuf);
    // softmax over L=16 + weighted sum -> d (new attns)
    softmax_attend<<<4096, 256, 0, stream>>>(sbuf, attn_states, dout + off_d);
    // output = [h_new|d] @ W2 + b2 ; also store into row 15 of new_attn_states
    gemm_bf16<1><<<dim3(4, 32), 256, 0, stream>>>(dout + off_h, dout + off_d, 512, 1024,
        W2at, W2bt, b2, dout + off_out, 512, dout + off_nas, 8192, 15 * 512,
        nullptr, nullptr, nullptr);
    // shift window: rows 0..14 of new_attn_states
    shift_states<<<g1((size_t)4096 * 1920), 256, 0, stream>>>(attn_states, dout + off_nas);
}
